// SpecSpatFTModel_31980326486475
// MI455X (gfx1250) — matrix operand order repaired and checked by reading, not compiled, not run on hardware
//
#include <hip/hip_runtime.h>

typedef __attribute__((ext_vector_type(16))) __bf16    v16bf;
typedef __attribute__((ext_vector_type(8)))  float     v8f;
typedef __attribute__((ext_vector_type(8)))  unsigned  v8u;

// ---------------- problem constants ----------------
#define NB   128      // batch
#define NC   30       // channels
#define NHW  128      // image H=W
#define FLATK 7680    // NC*16*16
#define HID  256
#define NCLS 3
#define KSTEPS1 240   // FLATK/32

// ---------------- split / pack helpers (bf16x3 emulated-fp32) ----------------
// Truncation split: f ~= hi + lo with |err| <~ 2^-16 |f| (lo captures hi's
// truncation residual, so RNE on hi buys nothing).
__device__ __forceinline__ void split_trunc(float f, unsigned short& hi, unsigned short& lo) {
    unsigned u = __builtin_bit_cast(unsigned, f);
    hi = (unsigned short)(u >> 16);
    float r = f - __builtin_bit_cast(float, u & 0xffff0000u);
    lo = (unsigned short)(__builtin_bit_cast(unsigned, r) >> 16);
}
// Split a pair of f32 and pack straight into fragment dwords:
// hid = bf16(f0) | bf16(f1)<<16 via one v_perm_b32 each.
__device__ __forceinline__ void split_pair(float f0, float f1, unsigned& hid, unsigned& lod) {
    unsigned u0 = __builtin_bit_cast(unsigned, f0);
    unsigned u1 = __builtin_bit_cast(unsigned, f1);
    hid = __builtin_amdgcn_perm(u1, u0, 0x07060302u);     // [u1.b3 u1.b2 u0.b3 u0.b2]
    float r0 = f0 - __builtin_bit_cast(float, u0 & 0xffff0000u);
    float r1 = f1 - __builtin_bit_cast(float, u1 & 0xffff0000u);
    lod = __builtin_amdgcn_perm(__builtin_bit_cast(unsigned, r1),
                                __builtin_bit_cast(unsigned, r0), 0x07060302u);
}
__device__ __forceinline__ __bf16 asbf(unsigned short u) {
    return __builtin_bit_cast(__bf16, u);
}
__device__ __forceinline__ v16bf ldfrag(const unsigned* p) {   // 32B fragment load (2x b128)
    return __builtin_bit_cast(v16bf, *(const v8u*)p);
}
__device__ __forceinline__ v8f wmma_bf16(v16bf a, v16bf b, v8f c) {
    return __builtin_amdgcn_wmma_f32_16x16x32_bf16(false, a, false, b, (short)0, c, false, false);
}

// WMMA fragment K-index maps (CDNA5 ISA 7.12.2, 16-bit operands, wave32)
__device__ __forceinline__ int kA16(int v, int halfk) {       // A: 16x32 (MxK)
    return (v >> 2) * 16 + halfk * 8 + (v & 3) * 2;
}
__device__ __forceinline__ int kB16(int v, int halfk) {       // B: 32x16 (KxN)
    return (v >> 2) * 16 + halfk * 8 + (v & 3) * 2;
}

// ---------------- workspace layout (fragment-packed dword images) ----------------
static constexpr size_t OFF_CROWH = 0;      // stage-1 A basis: 4 ksteps*32 lanes*8 dw = 4KB
static constexpr size_t OFF_CROWL = 4096;
static constexpr size_t OFF_SROWH = 8192;
static constexpr size_t OFF_SROWL = 12288;
static constexpr size_t OFF_CTBH  = 16384;  // stage-2 B basis, same shape
static constexpr size_t OFF_CTBL  = 20480;
static constexpr size_t OFF_STBH  = 24576;
static constexpr size_t OFF_STBL  = 28672;
static constexpr size_t H_DW      = (size_t)8  * KSTEPS1 * 32 * 8;   // 491520 dw
static constexpr size_t W1_DW     = (size_t)16 * KSTEPS1 * 32 * 8;   // 983040 dw
static constexpr size_t OFF_HH    = 32768;
static constexpr size_t OFF_HL    = OFF_HH  + H_DW * 4;
static constexpr size_t OFF_W1H   = OFF_HL  + H_DW * 4;
static constexpr size_t OFF_W1L   = OFF_W1H + W1_DW * 4;
static constexpr size_t OFF_HID   = OFF_W1L + W1_DW * 4;             // 128x256 f32

// ================= kernel 1: DFT basis, pre-packed in fragment order =================
__global__ void __launch_bounds__(256) basis_init_kernel(
    unsigned* CrowH, unsigned* CrowL, unsigned* SrowH, unsigned* SrowL,
    unsigned* CtbH,  unsigned* CtbL,  unsigned* StbH,  unsigned* StbL)
{
    const int tid = threadIdx.x;
    // stage-1 A fragments: A[p][h] = cos/sin(pi*(p-8)*h/64), p=lane&15
    for (int idx = tid; idx < 4 * 32 * 8; idx += 256) {
        int v = idx & 7, lane = (idx >> 3) & 31, ks = idx >> 8;
        int p = lane & 15, hf = lane >> 4;
        int h = ks * 32 + kA16(v, hf);
        float a0 = (float)((p - 8) * h)       * (1.0f / 64.0f);
        float a1 = (float)((p - 8) * (h + 1)) * (1.0f / 64.0f);
        unsigned dh, dl;
        split_pair(cospif(a0), cospif(a1), dh, dl); CrowH[idx] = dh; CrowL[idx] = dl;
        split_pair(sinpif(a0), sinpif(a1), dh, dl); SrowH[idx] = dh; SrowL[idx] = dl;
    }
    // stage-2 B fragments: B[w][q] = cos/sin(pi*(q-8)*w/64), q=lane&15
    for (int idx = tid; idx < 4 * 32 * 8; idx += 256) {
        int v = idx & 7, lane = (idx >> 3) & 31, ks = idx >> 8;
        int q = lane & 15, hf = lane >> 4;
        int w = ks * 32 + kB16(v, hf);
        float a0 = (float)((q - 8) * w)       * (1.0f / 64.0f);
        float a1 = (float)((q - 8) * (w + 1)) * (1.0f / 64.0f);
        unsigned dh, dl;
        split_pair(cospif(a0), cospif(a1), dh, dl); CtbH[idx] = dh; CtbL[idx] = dl;
        split_pair(sinpif(a0), sinpif(a1), dh, dl); StbH[idx] = dh; StbL[idx] = dl;
    }
}

// ================= kernel 2: W1 -> split bf16, packed as GEMM B fragments =================
// dword index = (((tn*240 + ks)*32) + lane)*8 + v ; B[k][n] = W1[n][k]
__global__ void __launch_bounds__(256) w1_pack_kernel(
    const float* __restrict__ W1, unsigned* __restrict__ W1h, unsigned* __restrict__ W1l)
{
    size_t idx = (size_t)blockIdx.x * 256 + threadIdx.x;
    if (idx >= W1_DW) return;
    int v = idx & 7;
    size_t t = idx >> 3;
    int lane = t & 31;
    size_t t2 = t >> 5;
    int ks = (int)(t2 % KSTEPS1), tn = (int)(t2 / KSTEPS1);
    int hf = lane >> 4, lm = lane & 15;
    int n = tn * 16 + lm;
    int k = ks * 32 + kB16(v, hf);
    unsigned dh, dl;
    split_pair(W1[(size_t)n * FLATK + k], W1[(size_t)n * FLATK + k + 1], dh, dl);
    W1h[idx] = dh; W1l[idx] = dl;
}

// ================= kernel 3: pruned 16x16 DFT + log-magnitude =================
// One block per (b,c) image; 8 waves; x read from HBM exactly once (coalesced f32).
__global__ void __launch_bounds__(256) dft16_kernel(
    const float* __restrict__ x, const float* __restrict__ fw,
    const unsigned* __restrict__ CrowH, const unsigned* __restrict__ CrowL,
    const unsigned* __restrict__ SrowH, const unsigned* __restrict__ SrowL,
    const unsigned* __restrict__ CtbH,  const unsigned* __restrict__ CtbL,
    const unsigned* __restrict__ StbH,  const unsigned* __restrict__ StbL,
    unsigned* __restrict__ hH, unsigned* __restrict__ hL)
{
    __shared__ unsigned short sTcHi[16 * 128], sTcLo[16 * 128];
    __shared__ unsigned short sTsHi[16 * 128], sTsLo[16 * 128];
    __shared__ float sP[8][256];

    const int img  = blockIdx.x;            // 0..3839
    const int b    = img / NC, c = img % NC;
    const int tid  = threadIdx.x;
    const int wv   = tid >> 5;              // wave id 0..7 == w-tile
    const int lane = tid & 31;
    const int hf   = lane >> 4;
    const int lm   = lane & 15;
    const float* xim = x + (size_t)img * (NHW * NHW);

    // ---- stage 1: Tc = C*X, Ts = S*X (dual accumulators break WMMA RAW chains) ----
    v8f zero = {0.f, 0.f, 0.f, 0.f, 0.f, 0.f, 0.f, 0.f};
    v8f tcA = zero, tcB = zero, tsA = zero, tsB = zero;
    const int wcol = wv * 16 + lm;
    #pragma unroll
    for (int ks = 0; ks < 4; ++ks) {
        const int kk = ks * 32;
        v8u bh, bl;
        #pragma unroll
        for (int v = 0; v < 8; ++v) {       // B frag from x: split+pack via v_perm
            int h0 = kk + kB16(v, hf);
            unsigned dh, dl;
            split_pair(xim[h0 * NHW + wcol], xim[(h0 + 1) * NHW + wcol], dh, dl);
            bh[v] = dh; bl[v] = dl;
        }
        v16bf Bhi = __builtin_bit_cast(v16bf, bh);
        v16bf Blo = __builtin_bit_cast(v16bf, bl);
        const int fo = (ks * 32 + lane) * 8;     // 32B-aligned fragment offset
        v16bf Chi = ldfrag(CrowH + fo), Clo = ldfrag(CrowL + fo);
        v16bf Shi = ldfrag(SrowH + fo), Slo = ldfrag(SrowL + fo);
        tcA = wmma_bf16(Chi, Bhi, tcA);          // interleave Tc/Ts chains
        tsA = wmma_bf16(Shi, Bhi, tsA);
        tcB = wmma_bf16(Chi, Blo, tcB);
        tsB = wmma_bf16(Shi, Blo, tsB);
        tcB = wmma_bf16(Clo, Bhi, tcB);
        tsB = wmma_bf16(Slo, Bhi, tsB);
    }
    v8f accTc = tcA + tcB, accTs = tsA + tsB;
    // spill Tc/Ts tiles to LDS as split bf16 (stage-2 A operands)
    #pragma unroll
    for (int r = 0; r < 8; ++r) {
        int m = r + hf * 8;
        unsigned short uh, ul;
        split_trunc(accTc[r], uh, ul); sTcHi[m * 128 + wcol] = uh; sTcLo[m * 128 + wcol] = ul;
        split_trunc(accTs[r], uh, ul); sTsHi[m * 128 + wcol] = uh; sTsLo[m * 128 + wcol] = ul;
    }
    __syncthreads();

    // ---- stage 2: P1=Tc*Ct, P2=Ts*St, P3=Ts*Ct, P4=Tc*St (2 waves/chain, K split) ----
    const int chain = wv >> 1;
    const int ksel  = wv & 1;
    const unsigned short* Ah = (chain == 0 || chain == 3) ? sTcHi : sTsHi;
    const unsigned short* Al = (chain == 0 || chain == 3) ? sTcLo : sTsLo;
    const unsigned*       Bh = (chain == 0 || chain == 2) ? CtbH : StbH;
    const unsigned*       Bl = (chain == 0 || chain == 2) ? CtbL : StbL;
    v8f pA = zero, pB = zero;
    #pragma unroll
    for (int kq = 0; kq < 2; ++kq) {
        int ks = ksel * 2 + kq;
        int kk = ks * 32;
        v16bf Ahi, Alo;
        #pragma unroll
        for (int v = 0; v < 8; ++v) {
            int ia = lm * 128 + kk + kA16(v, hf);
            Ahi[2 * v] = asbf(Ah[ia]); Ahi[2 * v + 1] = asbf(Ah[ia + 1]);
            Alo[2 * v] = asbf(Al[ia]); Alo[2 * v + 1] = asbf(Al[ia + 1]);
        }
        const int fo = (ks * 32 + lane) * 8;
        v16bf Bhi = ldfrag(Bh + fo), Blo = ldfrag(Bl + fo);
        pA = wmma_bf16(Ahi, Bhi, pA);
        pB = wmma_bf16(Ahi, Blo, pB);
        pB = wmma_bf16(Alo, Bhi, pB);
    }
    v8f acc = pA + pB;
    #pragma unroll
    for (int r = 0; r < 8; ++r)
        sP[wv][(r + hf * 8) * 16 + lm] = acc[r];
    __syncthreads();

    // ---- finalize: Fr = P1-P2, |Fi| = P3+P4; h = log(|F|+1e-6)*fw ----
    float P1 = sP[0][tid] + sP[1][tid];
    float P2 = sP[2][tid] + sP[3][tid];
    float P3 = sP[4][tid] + sP[5][tid];
    float P4 = sP[6][tid] + sP[7][tid];
    float g1 = P1 - P2, g2 = P3 + P4;
    float mag = logf(sqrtf(g1 * g1 + g2 * g2) + 1e-6f) * fw[c * 256 + tid];
    // store straight into GEMM A-fragment order: m=b, k=c*256+tid
    unsigned short uh, ul;
    split_trunc(mag, uh, ul);
    int kg  = c * 256 + tid;
    int ks  = kg >> 5, kr = kg & 31;
    int g   = kr >> 4, rem = kr & 15;
    int hf2 = rem >> 3, vi = (rem & 7) >> 1, par = kr & 1;
    int v   = g * 4 + vi;
    int lane2 = hf2 * 16 + (b & 15);
    size_t dw = (((size_t)(b >> 4) * KSTEPS1 + ks) * 32 + lane2) * 8 + v;
    ((unsigned short*)hH)[dw * 2 + par] = uh;
    ((unsigned short*)hL)[dw * 2 + par] = ul;
}

// ================= kernel 4: hidden = h(128x7680) @ W1^T =================
// 128 tiles of 16x16, one wave per tile; all operands fragment-packed -> pure b128+WMMA loop.
__global__ void __launch_bounds__(256) gemm1_kernel(
    const unsigned* __restrict__ hH, const unsigned* __restrict__ hL,
    const unsigned* __restrict__ W1h, const unsigned* __restrict__ W1l,
    float* __restrict__ hid)
{
    const int tid  = threadIdx.x;
    const int wv   = tid >> 5;
    const int lane = tid & 31;
    const int hf   = lane >> 4;
    const int lm   = lane & 15;
    const int t    = blockIdx.x * 8 + wv;   // 0..127
    const int tm   = t >> 4, tn = t & 15;

    const unsigned* aH = hH  + ((size_t)tm * KSTEPS1 * 32 + lane) * 8;
    const unsigned* aL = hL  + ((size_t)tm * KSTEPS1 * 32 + lane) * 8;
    const unsigned* bH = W1h + ((size_t)tn * KSTEPS1 * 32 + lane) * 8;
    const unsigned* bL = W1l + ((size_t)tn * KSTEPS1 * 32 + lane) * 8;

    v8f zero = {0.f, 0.f, 0.f, 0.f, 0.f, 0.f, 0.f, 0.f};
    v8f accA = zero, accB = zero;
    for (int ks = 0; ks < KSTEPS1; ++ks) {
        v16bf Ahi = ldfrag(aH), Alo = ldfrag(aL);
        v16bf Bhi = ldfrag(bH), Blo = ldfrag(bL);
        accA = wmma_bf16(Ahi, Bhi, accA);
        accB = wmma_bf16(Ahi, Blo, accB);
        accB = wmma_bf16(Alo, Bhi, accB);
        aH += 256; aL += 256; bH += 256; bL += 256;   // 32 lanes * 8 dw
    }
    v8f acc = accA + accB;
    #pragma unroll
    for (int r = 0; r < 8; ++r)
        hid[(size_t)(tm * 16 + r + hf * 8) * HID + tn * 16 + lm] = acc[r];
}

// ================= kernel 5: bias + LayerNorm + ReLU + 3-class head =================
__global__ void __launch_bounds__(256) head_kernel(
    const float* __restrict__ hid, const float* __restrict__ b1,
    const float* __restrict__ gam, const float* __restrict__ bet,
    const float* __restrict__ W2,  const float* __restrict__ b2,
    float* __restrict__ out)
{
    __shared__ float red[256];
    const int b = blockIdx.x, j = threadIdx.x;
    float v = hid[b * HID + j] + b1[j];

    red[j] = v; __syncthreads();
    for (int s = 128; s > 0; s >>= 1) { if (j < s) red[j] += red[j + s]; __syncthreads(); }
    float mu = red[0] * (1.0f / 256.0f);
    __syncthreads();

    float d = v - mu;
    red[j] = d * d; __syncthreads();
    for (int s = 128; s > 0; s >>= 1) { if (j < s) red[j] += red[j + s]; __syncthreads(); }
    float var = red[0] * (1.0f / 256.0f);
    __syncthreads();

    float r = fmaxf(d * rsqrtf(var + 1e-5f) * gam[j] + bet[j], 0.0f);
    for (int cc = 0; cc < NCLS; ++cc) {
        red[j] = r * W2[cc * HID + j]; __syncthreads();
        for (int s = 128; s > 0; s >>= 1) { if (j < s) red[j] += red[j + s]; __syncthreads(); }
        if (j == 0) out[b * NCLS + cc] = red[0] + b2[cc];
        __syncthreads();
    }
}

// ================= launcher =================
extern "C" void kernel_launch(void* const* d_in, const int* in_sizes, int n_in,
                              void* d_out, int out_size, void* d_ws, size_t ws_size,
                              hipStream_t stream) {
    (void)in_sizes; (void)n_in; (void)out_size; (void)ws_size;
    const float* x   = (const float*)d_in[0];
    const float* fw  = (const float*)d_in[1];
    const float* W1  = (const float*)d_in[2];
    const float* b1  = (const float*)d_in[3];
    const float* gam = (const float*)d_in[4];
    const float* bet = (const float*)d_in[5];
    const float* W2  = (const float*)d_in[6];
    const float* b2  = (const float*)d_in[7];
    float* out = (float*)d_out;

    char* w = (char*)d_ws;
    unsigned* CrowH = (unsigned*)(w + OFF_CROWH);
    unsigned* CrowL = (unsigned*)(w + OFF_CROWL);
    unsigned* SrowH = (unsigned*)(w + OFF_SROWH);
    unsigned* SrowL = (unsigned*)(w + OFF_SROWL);
    unsigned* CtbH  = (unsigned*)(w + OFF_CTBH);
    unsigned* CtbL  = (unsigned*)(w + OFF_CTBL);
    unsigned* StbH  = (unsigned*)(w + OFF_STBH);
    unsigned* StbL  = (unsigned*)(w + OFF_STBL);
    unsigned* hH    = (unsigned*)(w + OFF_HH);
    unsigned* hL    = (unsigned*)(w + OFF_HL);
    unsigned* W1h   = (unsigned*)(w + OFF_W1H);
    unsigned* W1l   = (unsigned*)(w + OFF_W1L);
    float*    hid   = (float*)(w + OFF_HID);

    basis_init_kernel<<<1, 256, 0, stream>>>(CrowH, CrowL, SrowH, SrowL,
                                             CtbH, CtbL, StbH, StbL);
    w1_pack_kernel<<<(int)((W1_DW + 255) / 256), 256, 0, stream>>>(W1, W1h, W1l);
    dft16_kernel<<<NB * NC, 256, 0, stream>>>(x, fw, CrowH, CrowL, SrowH, SrowL,
                                              CtbH, CtbL, StbH, StbL, hH, hL);
    gemm1_kernel<<<16, 256, 0, stream>>>(hH, hL, W1h, W1l, hid);
    head_kernel<<<NB, 256, 0, stream>>>(hid, b1, gam, bet, W2, b2, out);
}
